// MultiHeadAttention_32384053411898
// MI455X (gfx1250) — matrix operand order repaired and checked by reading, not compiled, not run on hardware
//
#include <hip/hip_runtime.h>

// ---------------------------------------------------------------------------
// MHA for MI455X (gfx1250, wave32, WMMA 16x16x32 f16 -> f32 accum)
// B=4, T=2048, D=1024, H=16, hd=64
// ---------------------------------------------------------------------------

typedef __attribute__((ext_vector_type(16))) _Float16 v16h;
typedef __attribute__((ext_vector_type(8)))  float    v8f;

#define DIM   1024
#define SEQ   2048
#define BATCH 4
#define HEADS 16
#define HDIM  64

#define LDG   88   // GEMM LDS row stride in halves (176 B: 16 B-aligned,
                   // gcd(44,64)=4 -> 16 consecutive rows conflict-free)
#define LDK   88   // attention LDS row stride in halves (same property)

union Frag {
    v16h h;
    uint4 q[2];
};

__device__ __forceinline__ v8f wmma_f16(v16h a, v16h b, v8f c) {
    return __builtin_amdgcn_wmma_f32_16x16x32_f16(
        /*neg_a=*/false, a, /*neg_b=*/false, b,
        /*c_mod=*/(short)0, c, /*reuse_a=*/false, /*reuse_b=*/false);
}

// LDS byte offset of a generic pointer to __shared__ memory (addrspace(3))
__device__ __forceinline__ unsigned lds_off(const void* p) {
    return (unsigned)(size_t)(const __attribute__((address_space(3))) void*)p;
}

// Async DMA: 16 bytes per lane, global -> LDS, tracked by ASYNCcnt
__device__ __forceinline__ void async_b128(unsigned dstLds, const _Float16* src) {
    asm volatile("global_load_async_to_lds_b128 %0, %1, off"
                 :: "v"(dstLds), "v"((unsigned long long)(size_t)src)
                 : "memory");
}

__device__ __forceinline__ void wait_async0() {
    asm volatile("s_wait_asynccnt 0x0" ::: "memory");
}

// ---------------------------------------------------------------------------
// Weight prep: Wt[n][k] = (f16) W[k][n]  for the four 1024x1024 weights
// ---------------------------------------------------------------------------
__global__ __launch_bounds__(256)
void transpose_w(const float* __restrict__ w0, const float* __restrict__ w1,
                 const float* __restrict__ w2, const float* __restrict__ w3,
                 _Float16* __restrict__ t0, _Float16* __restrict__ t1,
                 _Float16* __restrict__ t2, _Float16* __restrict__ t3)
{
    __shared__ float tile[32][33];
    const float* W; _Float16* Wt;
    switch (blockIdx.z) {
        case 0:  W = w0; Wt = t0; break;
        case 1:  W = w1; Wt = t1; break;
        case 2:  W = w2; Wt = t2; break;
        default: W = w3; Wt = t3; break;
    }
    const int tx = threadIdx.x, ty = threadIdx.y;
    const int x = blockIdx.x * 32 + tx;      // n (input col)
    const int y = blockIdx.y * 32 + ty;      // k (input row)
#pragma unroll
    for (int j = 0; j < 32; j += 8)
        tile[ty + j][tx] = W[(size_t)(y + j) * DIM + x];   // tile[k_local][n_local]
    __syncthreads();
    const int x2 = blockIdx.y * 32 + tx;     // k (output col)
    const int y2 = blockIdx.x * 32 + ty;     // n (output row)
#pragma unroll
    for (int j = 0; j < 32; j += 8)
        Wt[(size_t)(y2 + j) * DIM + x2] = (_Float16)tile[tx][ty + j];
}

// ---------------------------------------------------------------------------
// GEMM: out[M=8192][N=1024] = A[8192][1024] * Wt^T + bias
//   Wt stored as [N][K] f16 (pre-transposed).
//   AT     = float (raw inputs) or _Float16 (attention output)
//   OMODE  = 0: f16 row-major out   (Q, K projections)
//            1: f16 out scattered as Vt[B][H][hd][T]  (V projection)
//            2: f32 row-major out   (final output projection)
// 128x128x64 tiles, 8 waves, wave tile = 32x64, register-staged prefetch:
//   storeLDS(cur) -> barrier -> issue global loads(next) -> 16 WMMAs -> barrier
// ---------------------------------------------------------------------------
template <typename AT, int OMODE>
__global__ __launch_bounds__(256)
void gemm_bias(const AT* __restrict__ A, const _Float16* __restrict__ Wt,
               const float* __restrict__ bias, void* __restrict__ out)
{
    __shared__ alignas(16) _Float16 lA[128 * LDG];
    __shared__ alignas(16) _Float16 lB[128 * LDG];

    const int tid  = threadIdx.x;
    const int lane = tid & 31;
    const int wave = tid >> 5;
    const int lh   = lane & 15;
    const int hi   = lane >> 4;            // 0 | 1
    const int a0   = hi ? 8 : 0;           // A-fragment K sub-offset

    const int m0 = blockIdx.y * 128;
    const int n0 = blockIdx.x * 128;
    const int wm = (wave & 3) * 32;        // wave row offset in tile
    const int wn = (wave >> 2) * 64;       // wave col offset in tile

    v8f acc[2][4] = {};

    // staging: 2 threads/row, 32 halves each (128 rows x 64 K)
    const int sRow = tid >> 1;
    const int sOff = (tid & 1) << 5;       // 0 or 32 elements

    const AT*       aSrc = A  + (size_t)(m0 + sRow) * DIM + sOff;
    const _Float16* bSrc = Wt + (size_t)(n0 + sRow) * DIM + sOff;

    uint4 ra[4], rb[4];
    auto loadA = [&](int k0) {
        if constexpr (sizeof(AT) == 4) {
            const float4* s = (const float4*)((const float*)aSrc + k0);
            float f[32];
#pragma unroll
            for (int i = 0; i < 8; ++i) ((float4*)f)[i] = s[i];
            _Float16 hb[32];
#pragma unroll
            for (int i = 0; i < 32; ++i) hb[i] = (_Float16)f[i];
#pragma unroll
            for (int i = 0; i < 4; ++i) ra[i] = ((uint4*)hb)[i];
        } else {
            const uint4* s = (const uint4*)((const _Float16*)aSrc + k0);
#pragma unroll
            for (int i = 0; i < 4; ++i) ra[i] = s[i];
        }
    };
    auto loadB = [&](int k0) {
        const uint4* s = (const uint4*)(bSrc + k0);
#pragma unroll
        for (int i = 0; i < 4; ++i) rb[i] = s[i];
    };

    loadA(0);
    loadB(0);

    for (int k0 = 0; k0 < DIM; k0 += 64) {
        // ---- commit staged registers to LDS ----
        {
            uint4* da = (uint4*)&lA[sRow * LDG + sOff];
            uint4* db = (uint4*)&lB[sRow * LDG + sOff];
#pragma unroll
            for (int i = 0; i < 4; ++i) da[i] = ra[i];
#pragma unroll
            for (int i = 0; i < 4; ++i) db[i] = rb[i];
        }
        __syncthreads();

        // ---- issue next tile's global loads (latency hidden by WMMAs) ----
        if (k0 + 64 < DIM) {
            loadA(k0 + 64);
            loadB(k0 + 64);
        }

        // ---- 2 K-chunks x (2x4) WMMAs from LDS ----
#pragma unroll
        for (int c = 0; c < 2; ++c) {
            Frag af[2], bf[4];
#pragma unroll
            for (int mt = 0; mt < 2; ++mt) {
                const _Float16* p = &lA[(wm + mt * 16 + lh) * LDG + c * 32 + a0];
                af[mt].q[0] = *(const uint4*)p;          // K = a0..a0+7
                af[mt].q[1] = *(const uint4*)(p + 16);   // K = a0+16..a0+23
            }
#pragma unroll
            for (int nt = 0; nt < 4; ++nt) {
                const _Float16* p = &lB[(wn + nt * 16 + lh) * LDG + c * 32 + a0];
                bf[nt].q[0] = *(const uint4*)p;
                bf[nt].q[1] = *(const uint4*)(p + 16);
            }
#pragma unroll
            for (int mt = 0; mt < 2; ++mt)
#pragma unroll
                for (int nt = 0; nt < 4; ++nt)
                    acc[mt][nt] = wmma_f16(af[mt].h, bf[nt].h, acc[mt][nt]);
        }
        __syncthreads();
    }

    // ---- epilogue: bias + store ----
#pragma unroll
    for (int mt = 0; mt < 2; ++mt)
#pragma unroll
        for (int nt = 0; nt < 4; ++nt) {
            const int col = n0 + wn + nt * 16 + lh;
            const float bv = bias[col];
            const int rowb = m0 + wm + mt * 16 + hi * 8;
#pragma unroll
            for (int r = 0; r < 8; ++r) {
                const int row = rowb + r;
                const float vv = acc[mt][nt][r] + bv;
                if constexpr (OMODE == 0) {
                    ((_Float16*)out)[(size_t)row * DIM + col] = (_Float16)vv;
                } else if constexpr (OMODE == 2) {
                    ((float*)out)[(size_t)row * DIM + col] = vv;
                } else {
                    // Vt[b][h][hd][t]
                    const int bb = row >> 11, t = row & (SEQ - 1);
                    const int hh = col >> 6,  hd = col & (HDIM - 1);
                    ((_Float16*)out)[(((size_t)bb * HEADS + hh) * HDIM + hd) * SEQ + t]
                        = (_Float16)vv;
                }
            }
        }
}

// ---------------------------------------------------------------------------
// Flash attention: 1 WG = 128 queries of one (b,h); each wave owns 16 rows.
// 64-key blocks. K/V tiles are DOUBLE-BUFFERED and filled with
// GLOBAL_LOAD_ASYNC_TO_LDS_B128 (ASYNCcnt) so the next tile's DMA overlaps
// the current tile's WMMAs. Row-sum reduction deferred to one end pass.
//   Qg, Kg : [B*T][D] f16 (head h at column offset h*64)
//   Vt     : [B][H][hd][T] f16
//   Og     : [B*T][D] f16 attention output (pre-W_o)
// ---------------------------------------------------------------------------
__global__ __launch_bounds__(256)
void attention(const _Float16* __restrict__ Qg, const _Float16* __restrict__ Kg,
               const _Float16* __restrict__ Vt, _Float16* __restrict__ Og)
{
    __shared__ alignas(16) _Float16 lK[2][64 * LDK];     // K tile  [key][hd]
    __shared__ alignas(16) _Float16 lV[2][64 * LDK];     // Vt tile [hd][key]
    __shared__ alignas(16) _Float16 lP[8 * 16 * LDK];    // per-wave 16x64 P

    const int tid  = threadIdx.x;
    const int lane = tid & 31;
    const int wave = tid >> 5;
    const int lh   = lane & 15;
    const int hi   = lane >> 4;
    const int a0   = hi ? 8 : 0;

    const int bid  = blockIdx.x;           // b*256 + h*16 + qblk
    const int qblk = bid & 15;
    const int h    = (bid >> 4) & 15;
    const int b    = bid >> 8;
    const int q0   = qblk * 128 + wave * 16;
    const size_t rowQ = (size_t)b * SEQ + q0;

    // exp2-domain scale: 1/sqrt(64) * log2(e)
    const float SC2 = 0.125f * 1.4426950408889634f;

    // ---- Q fragments (persist in registers, cover hd 0..63) ----
    Frag qf[2];
    {
        const _Float16* qp = Qg + (rowQ + lh) * DIM + h * HDIM;
#pragma unroll
        for (int c = 0; c < 2; ++c) {
            qf[c].q[0] = *(const uint4*)(qp + c * 32 + a0);
            qf[c].q[1] = *(const uint4*)(qp + c * 32 + a0 + 16);
        }
    }

    v8f o[4] = {};
    float mrow[8], lrow[8];                // lrow = per-lane PARTIAL row sums
#pragma unroll
    for (int r = 0; r < 8; ++r) { mrow[r] = -INFINITY; lrow[r] = 0.0f; }

    const _Float16* kbase = Kg + (size_t)b * SEQ * DIM + h * HDIM;
    const _Float16* vbase = Vt + (((size_t)b * HEADS + h) * HDIM) * SEQ;
    _Float16* pP = &lP[wave * 16 * LDK];

    // staging coords: 4 threads/row, 16 halves (= 2 async b128) each
    const int sRow = tid >> 2;             // 0..63
    const int sOff = (tid & 3) << 4;       // 0,16,32,48

    // per-thread LDS destination offsets (bytes) for both buffers
    const unsigned kDst0 = lds_off(&lK[0][sRow * LDK + sOff]);
    const unsigned kDst1 = lds_off(&lK[1][sRow * LDK + sOff]);
    const unsigned vDst0 = lds_off(&lV[0][sRow * LDK + sOff]);
    const unsigned vDst1 = lds_off(&lV[1][sRow * LDK + sOff]);

    auto stage = [&](int buf, int kb) {
        const _Float16* ks = kbase + (size_t)(kb + sRow) * DIM + sOff;
        const _Float16* vs = vbase + (size_t)sRow * SEQ + kb + sOff;
        const unsigned kd = buf ? kDst1 : kDst0;
        const unsigned vd = buf ? vDst1 : vDst0;
        async_b128(kd,      ks);
        async_b128(kd + 16, ks + 8);
        async_b128(vd,      vs);
        async_b128(vd + 16, vs + 8);
    };

    stage(0, 0);

    for (int it = 0; it < SEQ / 64; ++it) {
        const int kb  = it * 64;
        const int cur = it & 1;

        wait_async0();          // my DMA for buffer `cur` complete
        __syncthreads();        // everyone's DMA complete; prev reads of other
                                // buffer are also done (program order)
        if (it + 1 < SEQ / 64) stage(cur ^ 1, kb + 64);

        const _Float16* K = lK[cur];
        const _Float16* V = lV[cur];

        // ---- scores S[16q][64k] : 4 tiles x 2 hd-chunks ----
        v8f s[4] = {};
#pragma unroll
        for (int nt = 0; nt < 4; ++nt)
#pragma unroll
            for (int c = 0; c < 2; ++c) {
                Frag kf;
                const _Float16* p = &K[(nt * 16 + lh) * LDK + c * 32 + a0];
                kf.q[0] = *(const uint4*)p;
                kf.q[1] = *(const uint4*)(p + 16);
                s[nt] = wmma_f16(qf[c].h, kf.h, s[nt]);
            }

        // ---- online softmax: row-MAX reduction, step-major for ILP ----
        float v[8];
#pragma unroll
        for (int r = 0; r < 8; ++r)
            v[r] = fmaxf(fmaxf(s[0][r], s[1][r]), fmaxf(s[2][r], s[3][r]));
#pragma unroll
        for (int st = 1; st < 16; st <<= 1)
#pragma unroll
            for (int r = 0; r < 8; ++r)
                v[r] = fmaxf(v[r], __shfl_xor(v[r], st));

        float corr[8];
#pragma unroll
        for (int r = 0; r < 8; ++r) {
            const float mnew = fmaxf(mrow[r], v[r]);
            corr[r] = exp2f((mrow[r] - mnew) * SC2);
            mrow[r] = mnew;
        }
#pragma unroll
        for (int r = 0; r < 8; ++r) {
            float ps = 0.0f;
#pragma unroll
            for (int nt = 0; nt < 4; ++nt) {
                const float p = exp2f((s[nt][r] - mrow[r]) * SC2);
                pP[(r + hi * 8) * LDK + nt * 16 + lh] = (_Float16)p;
                ps += p;
            }
            lrow[r] = lrow[r] * corr[r] + ps;   // per-lane partial sum
        }
#pragma unroll
        for (int t = 0; t < 4; ++t)
#pragma unroll
            for (int r = 0; r < 8; ++r) o[t][r] *= corr[r];

        // ---- P (C-layout) -> A-fragments via per-wave LDS tile ----
        Frag pf[2];
#pragma unroll
        for (int c = 0; c < 2; ++c) {
            const _Float16* p = pP + lh * LDK + c * 32 + a0;
            pf[c].q[0] = *(const uint4*)p;
            pf[c].q[1] = *(const uint4*)(p + 16);
        }
        // ---- O += P * V ----
#pragma unroll
        for (int nt = 0; nt < 4; ++nt)
#pragma unroll
            for (int c = 0; c < 2; ++c) {
                Frag vf;
                const _Float16* p = &V[(nt * 16 + lh) * LDK + c * 32 + a0];
                vf.q[0] = *(const uint4*)p;
                vf.q[1] = *(const uint4*)(p + 16);
                o[nt] = wmma_f16(pf[c].h, vf.h, o[nt]);
            }
    }

    // ---- single deferred row-sum reduction (step-major), normalize, store --
#pragma unroll
    for (int st = 1; st < 16; st <<= 1)
#pragma unroll
        for (int r = 0; r < 8; ++r)
            lrow[r] += __shfl_xor(lrow[r], st);
#pragma unroll
    for (int r = 0; r < 8; ++r) {
        const float inv = 1.0f / lrow[r];
        const size_t grow = rowQ + hi * 8 + r;
#pragma unroll
        for (int t = 0; t < 4; ++t)
            Og[grow * DIM + h * HDIM + t * 16 + lh] = (_Float16)(o[t][r] * inv);
    }
}

// ---------------------------------------------------------------------------
// Launch
// ---------------------------------------------------------------------------
extern "C" void kernel_launch(void* const* d_in, const int* in_sizes, int n_in,
                              void* d_out, int out_size, void* d_ws, size_t ws_size,
                              hipStream_t stream)
{
    const float* q   = (const float*)d_in[0];
    const float* k   = (const float*)d_in[1];
    const float* v   = (const float*)d_in[2];
    const float* W_q = (const float*)d_in[3];
    const float* b_q = (const float*)d_in[4];
    const float* W_k = (const float*)d_in[5];
    const float* b_k = (const float*)d_in[6];
    const float* W_v = (const float*)d_in[7];
    const float* b_v = (const float*)d_in[8];
    const float* W_o = (const float*)d_in[9];
    const float* b_o = (const float*)d_in[10];

    char* ws = (char*)d_ws;
    size_t off = 0;
    auto carve = [&](size_t bytes) -> void* {
        void* p = ws + off;
        off += (bytes + 255) & ~(size_t)255;
        return p;
    };
    const size_t wbytes = (size_t)DIM * DIM * sizeof(_Float16);          // 2 MiB
    const size_t abytes = (size_t)BATCH * SEQ * DIM * sizeof(_Float16);  // 16 MiB
    _Float16* WtQ  = (_Float16*)carve(wbytes);
    _Float16* WtK  = (_Float16*)carve(wbytes);
    _Float16* WtV  = (_Float16*)carve(wbytes);
    _Float16* WtO  = (_Float16*)carve(wbytes);
    _Float16* Qh   = (_Float16*)carve(abytes);
    _Float16* Kh   = (_Float16*)carve(abytes);
    _Float16* Vth  = (_Float16*)carve(abytes);
    _Float16* attn = (_Float16*)carve(abytes);
    (void)ws_size; (void)in_sizes; (void)n_in; (void)out_size;

    // 1) weights -> f16, transposed [N][K]
    transpose_w<<<dim3(32, 32, 4), dim3(32, 8, 1), 0, stream>>>(
        W_q, W_k, W_v, W_o, WtQ, WtK, WtV, WtO);

    // 2) projections
    const dim3 gg(DIM / 128, (BATCH * SEQ) / 128);   // (8, 64)
    gemm_bias<float, 0><<<gg, 256, 0, stream>>>(q, WtQ, b_q, Qh);
    gemm_bias<float, 0><<<gg, 256, 0, stream>>>(k, WtK, b_k, Kh);
    gemm_bias<float, 1><<<gg, 256, 0, stream>>>(v, WtV, b_v, Vth);

    // 3) flash attention: B*H*(T/128) = 1024 workgroups
    attention<<<BATCH * HEADS * (SEQ / 128), 256, 0, stream>>>(Qh, Kh, Vth, attn);

    // 4) output projection -> fp32
    gemm_bias<_Float16, 2><<<gg, 256, 0, stream>>>(attn, WtO, b_o, (float*)d_out);
}
